// PatchAttentionWithPairBias_44126493999686
// MI455X (gfx1250) — matrix operand order repaired and checked by reading, not compiled, not run on hardware
//
#include <hip/hip_runtime.h>
#include <hip/hip_bf16.h>
#include <stdint.h>

// ---------------------------------------------------------------------------
// PatchAttentionWithPairBias on MI455X (gfx1250, wave32, WMMA)
//
// B=4 N=512 Df=256 Dp=128 A=256 H=8 hd=32
// Roofline: dominated by streaming D (512 MB) once -> ~23 us at 23.3 TB/s.
// All GEMMs (including the dominant LN(D)@Wp) use v_wmma_f32_16x16x32_bf16.
//
// Pair-path refactor:  LN(x)*g @ Wp + b@Wp  ==  rstd*(x@W' - mean*S) + bk
// with W' = diag(g_pair)*Wp (bf16), S_k = sum_i W'[i,k], bk_k = sum_i b_i*Wp[i,k]
// all precomputed once, so the 512 MB stream needs no per-element normalize:
// raw x is packed to bf16 (v_perm_b32 per pair) and fed straight to WMMA.
// ---------------------------------------------------------------------------

typedef __attribute__((ext_vector_type(16))) __bf16 v16bf;
typedef __attribute__((ext_vector_type(8)))  float  v8f;
typedef __attribute__((ext_vector_type(4)))  float  f4;
typedef __attribute__((ext_vector_type(4)))  unsigned int u4;

__device__ __forceinline__ unsigned short f2bf(float f) {
    unsigned int x = __float_as_uint(f);
    unsigned int r = x + 0x7fffu + ((x >> 16) & 1u);   // round-to-nearest-even
    return (unsigned short)(r >> 16);
}
// pack two fp32 -> packed bf16 dword (lo in [15:0]): one add each + v_perm_b32
__device__ __forceinline__ unsigned int pkbf(float lo, float hi) {
    unsigned int a = __float_as_uint(lo) + 0x8000u;
    unsigned int b = __float_as_uint(hi) + 0x8000u;
    return __builtin_amdgcn_perm(b, a, 0x07060302u);   // {b.byte3,b.byte2,a.byte3,a.byte2}
}
__device__ __forceinline__ float wred_sum(float v) {
    #pragma unroll
    for (int m = 16; m > 0; m >>= 1) v += __shfl_xor(v, m, 32);
    return v;
}
__device__ __forceinline__ float wred_max(float v) {
    #pragma unroll
    for (int m = 16; m > 0; m >>= 1) v = fmaxf(v, __shfl_xor(v, m, 32));
    return v;
}

// ---------------------------------------------------------------------------
// K0: one-wave precompute of W' (bf16), S[8], bk[8] from Wp/g_pair/b_pair.
// ---------------------------------------------------------------------------
__global__ __launch_bounds__(32) void kp_prep(
    const float* __restrict__ Wp, const float* __restrict__ gp,
    const float* __restrict__ bp, unsigned short* __restrict__ wpb,
    float* __restrict__ Svec, float* __restrict__ Bkvec)
{
    const int lane = threadIdx.x;
    float S[8] = {0,0,0,0,0,0,0,0};
    float Bk[8] = {0,0,0,0,0,0,0,0};
    #pragma unroll
    for (int t = 0; t < 4; ++t) {
        const int i = lane * 4 + t;
        const float gi = gp[i], bi = bp[i];
        #pragma unroll
        for (int k = 0; k < 8; ++k) {
            const float w  = Wp[i * 8 + k];
            const float wp = gi * w;
            wpb[i * 8 + k] = f2bf(wp);
            S[k]  += wp;
            Bk[k] += bi * w;
        }
    }
    #pragma unroll
    for (int k = 0; k < 8; ++k) { S[k] = wred_sum(S[k]); Bk[k] = wred_sum(Bk[k]); }
    if (lane == 0) {
        #pragma unroll
        for (int k = 0; k < 8; ++k) { Svec[k] = S[k]; Bkvec[k] = Bk[k]; }
    }
}

// ---------------------------------------------------------------------------
// K1: pair pathway, WMMA version.  One wave per 16-row tile of D.
// Raw rows -> bf16 A-fragments (K=128, 4 WMMA steps) against constant W'
// B-fragments; row mean/rstd from one shfl_xor(16) (lane l / l+16 hold the
// two halves of row l%16, matching the A layout); 1KB LDS redistributes the
// stats to the D-fragment row order for the affine epilogue.
// ---------------------------------------------------------------------------
__global__ __launch_bounds__(256) void kp_pair(
    const float* __restrict__ Dm, const unsigned short* __restrict__ wpb,
    const float* __restrict__ Svec, const float* __restrict__ Bkvec,
    float* __restrict__ outp)
{
    __shared__ float smean[8][16];
    __shared__ float srstd[8][16];
    const int lane = threadIdx.x & 31;
    const int wv   = threadIdx.x >> 5;
    const int g    = lane >> 4;
    const int m    = lane & 15;

    // constant B fragments: B[kk][n] = W'[32s+kk][n], zero-padded cols 8..15
    v16bf Bf[4];
    #pragma unroll
    for (int s = 0; s < 4; ++s) {
        union { v16bf v; unsigned short h[16]; } t;
        #pragma unroll
        for (int e = 0; e < 16; ++e)
            t.h[e] = (m < 8) ? wpb[(32 * s + 8 * g + (e & 7) + 16 * (e >> 3)) * 8 + m] : (unsigned short)0;
        Bf[s] = t.v;
    }
    const float Sm = (m < 8) ? Svec[m]  : 0.f;
    const float Bm = (m < 8) ? Bkvec[m] : 0.f;

    const long long tile = (long long)blockIdx.x * 8 + wv;
    const float* rp = Dm + (tile * 16 + m) * 128;   // lane's A row = m

    v8f acc = {0.f,0.f,0.f,0.f,0.f,0.f,0.f,0.f};
    float sum = 0.f, ss = 0.f;
    #pragma unroll
    for (int s = 0; s < 4; ++s) {
        const float* p0 = rp + 32 * s + 8 * g;        // elems kk =  8g..8g+7
        const float* p1 = rp + 32 * s + 16 + 8 * g;   // elems kk = 16+8g..+7
        const f4 a0 = __builtin_nontemporal_load((const f4*)p0);
        const f4 a1 = __builtin_nontemporal_load((const f4*)(p0 + 4));
        const f4 a2 = __builtin_nontemporal_load((const f4*)p1);
        const f4 a3 = __builtin_nontemporal_load((const f4*)(p1 + 4));
        sum += a0.x+a0.y+a0.z+a0.w + a1.x+a1.y+a1.z+a1.w
             + a2.x+a2.y+a2.z+a2.w + a3.x+a3.y+a3.z+a3.w;
        ss  += a0.x*a0.x+a0.y*a0.y+a0.z*a0.z+a0.w*a0.w
             + a1.x*a1.x+a1.y*a1.y+a1.z*a1.z+a1.w*a1.w
             + a2.x*a2.x+a2.y*a2.y+a2.z*a2.z+a2.w*a2.w
             + a3.x*a3.x+a3.y*a3.y+a3.z*a3.z+a3.w*a3.w;
        union { v16bf v; unsigned int u[8]; } A;
        A.u[0] = pkbf(a0.x, a0.y);  A.u[1] = pkbf(a0.z, a0.w);
        A.u[2] = pkbf(a1.x, a1.y);  A.u[3] = pkbf(a1.z, a1.w);
        A.u[4] = pkbf(a2.x, a2.y);  A.u[5] = pkbf(a2.z, a2.w);
        A.u[6] = pkbf(a3.x, a3.y);  A.u[7] = pkbf(a3.z, a3.w);
        acc = __builtin_amdgcn_wmma_f32_16x16x32_bf16(
                  false, A.v, false, Bf[s], (short)0, acc, false, false);
    }

    // row stats: lane l and l+16 each summed half of row m
    sum += __shfl_xor(sum, 16, 32);
    ss  += __shfl_xor(ss, 16, 32);
    const float mean = sum * (1.0f / 128.0f);
    const float var  = ss * (1.0f / 128.0f) - mean * mean;
    const float rstd = rsqrtf(var + 1e-3f);
    smean[wv][m] = mean;          // both halves write identical values
    srstd[wv][m] = rstd;          // LDS is in-order within a wave: no barrier

    #pragma unroll
    for (int r = 0; r < 8; ++r) {
        const int rloc = r + 8 * g;                   // D-fragment row
        const float mr = smean[wv][rloc];
        const float rr = srstd[wv][rloc];
        if (m < 8)
            outp[(tile * 16 + rloc) * 8 + m] = rr * (acc[r] - mr * Sm) + Bm;
    }
}

// ---------------------------------------------------------------------------
// K2: feature LN.  One wave per row of F (256 floats) -> bf16 Fn.
// ---------------------------------------------------------------------------
__global__ __launch_bounds__(256) void kp_featln(
    const float* __restrict__ F, const float* __restrict__ gf,
    const float* __restrict__ bf, unsigned short* __restrict__ fnbf)
{
    const int lane = threadIdx.x & 31;
    const int row  = blockIdx.x * 8 + (threadIdx.x >> 5);   // 2048 rows
    const float* fr = F + row * 256 + lane * 8;
    const f4 a = *(const f4*)fr;
    const f4 b = *(const f4*)(fr + 4);

    float s  = a.x+a.y+a.z+a.w + b.x+b.y+b.z+b.w;
    float ss = a.x*a.x+a.y*a.y+a.z*a.z+a.w*a.w + b.x*b.x+b.y*b.y+b.z*b.z+b.w*b.w;
    s  = wred_sum(s);
    ss = wred_sum(ss);
    const float mean = s * (1.0f / 256.0f);
    const float var  = ss * (1.0f / 256.0f) - mean * mean;
    const float rstd = rsqrtf(var + 1e-3f);

    const f4 g0 = *(const f4*)(gf + lane * 8);
    const f4 g1 = *(const f4*)(gf + lane * 8 + 4);
    const f4 b0 = *(const f4*)(bf + lane * 8);
    const f4 b1 = *(const f4*)(bf + lane * 8 + 4);

    union { unsigned short s[8]; u4 u; } o;
    o.s[0] = f2bf((a.x-mean)*rstd*g0.x + b0.x);
    o.s[1] = f2bf((a.y-mean)*rstd*g0.y + b0.y);
    o.s[2] = f2bf((a.z-mean)*rstd*g0.z + b0.z);
    o.s[3] = f2bf((a.w-mean)*rstd*g0.w + b0.w);
    o.s[4] = f2bf((b.x-mean)*rstd*g1.x + b1.x);
    o.s[5] = f2bf((b.y-mean)*rstd*g1.y + b1.y);
    o.s[6] = f2bf((b.z-mean)*rstd*g1.z + b1.z);
    o.s[7] = f2bf((b.w-mean)*rstd*g1.w + b1.w);
    *(u4*)(fnbf + row * 256 + lane * 8) = o.u;
}

// ---------------------------------------------------------------------------
// K3: Q/K/V/G projections.  One wave per 16x16 output tile of one matrix.
// ---------------------------------------------------------------------------
__global__ __launch_bounds__(32) void kp_qkvg(
    const unsigned short* __restrict__ fnbf,
    const float* __restrict__ Wq, const float* __restrict__ Wk,
    const float* __restrict__ Wv, const float* __restrict__ Wg,
    const float* __restrict__ bg,
    unsigned short* __restrict__ qbf, unsigned short* __restrict__ kbf,
    float* __restrict__ vf, float* __restrict__ gf)
{
    const int lane = threadIdx.x;
    const int g    = lane >> 4;
    const int m    = lane & 15;
    const int row0 = blockIdx.x * 16;
    const int col0 = blockIdx.y * 16;
    const int z    = blockIdx.z;
    const float* W = (z == 0) ? Wq : (z == 1) ? Wk : (z == 2) ? Wv : Wg;

    v8f acc = {0.f,0.f,0.f,0.f,0.f,0.f,0.f,0.f};
    for (int ks = 0; ks < 256; ks += 32) {
        union { v16bf v; u4 u[2]; } A;
        const unsigned short* ar = fnbf + (row0 + m) * 256 + ks;
        A.u[0] = *(const u4*)(ar + 8 * g);
        A.u[1] = *(const u4*)(ar + 16 + 8 * g);

        union { v16bf v; unsigned short s[16]; } Bf;
        #pragma unroll
        for (int e = 0; e < 16; ++e)
            Bf.s[e] = f2bf(W[(ks + 8 * g + (e & 7) + 16 * (e >> 3)) * 256 + col0 + m]);

        acc = __builtin_amdgcn_wmma_f32_16x16x32_bf16(
                  false, A.v, false, Bf.v, (short)0, acc, false, false);
    }

    const int col = col0 + m;
    if (z == 0) {
        #pragma unroll
        for (int r = 0; r < 8; ++r)
            qbf[(row0 + r + 8 * g) * 256 + col] = f2bf(acc[r] * 0.17677669529663687f);
    } else if (z == 1) {
        #pragma unroll
        for (int r = 0; r < 8; ++r)
            kbf[(row0 + r + 8 * g) * 256 + col] = f2bf(acc[r]);
    } else if (z == 2) {
        #pragma unroll
        for (int r = 0; r < 8; ++r)
            vf[(row0 + r + 8 * g) * 256 + col] = acc[r];
    } else {
        const float bgc = bg[col];
        #pragma unroll
        for (int r = 0; r < 8; ++r)
            gf[(row0 + r + 8 * g) * 256 + col] = acc[r] + bgc;
    }
}

// ---------------------------------------------------------------------------
// K4: scores + softmax-diagonal.  One wave per (b, h, 16-row p-tile).
// scores tile = WMMA(Qr, Kr^T, C = bias tile) -> LDS, then row max /
// sum-exp / diagonal -> diag weight.  Head-split addressing folded in.
// ---------------------------------------------------------------------------
__global__ __launch_bounds__(32) void kp_scores(
    const unsigned short* __restrict__ qbf,
    const unsigned short* __restrict__ kbf,
    const float* __restrict__ bpb, float* __restrict__ dw)
{
    __shared__ float sc[16 * 516];             // +4 pad kills bank conflicts
    const int lane = threadIdx.x;
    const int g    = lane >> 4;
    const int m    = lane & 15;
    const int pt   = blockIdx.x;
    const int h    = blockIdx.y;
    const int b    = blockIdx.z;

    const int n = pt * 16 + m;
    const unsigned short* qr = qbf + ((b * 512 + h * 64 + (n >> 3)) * 256) + (n & 7) * 32;
    union { v16bf v; u4 u[2]; } A;
    A.u[0] = *(const u4*)(qr + 8 * g);
    A.u[1] = *(const u4*)(qr + 16 + 8 * g);

    const float* bpbase = bpb + (size_t)(b * 8 + h) * 262144;

    for (int qt = 0; qt < 32; ++qt) {
        const int q = qt * 16 + m;
        const unsigned short* kr =
            kbf + ((b * 512 + h * 64 + (q >> 3)) * 256) + (q & 7) * 32 + 8 * g;
        union { v16bf v; u4 u[2]; } Bf;
        Bf.u[0] = *(const u4*)kr;
        Bf.u[1] = *(const u4*)(kr + 16);

        v8f c;
        #pragma unroll
        for (int r = 0; r < 8; ++r)
            c[r] = bpbase[(size_t)(pt * 16 + r + 8 * g) * 512 + qt * 16 + m];

        v8f d = __builtin_amdgcn_wmma_f32_16x16x32_bf16(
                    false, A.v, false, Bf.v, (short)0, c, false, false);

        #pragma unroll
        for (int r = 0; r < 8; ++r)
            sc[(r + 8 * g) * 516 + qt * 16 + m] = d[r];
    }
    __syncthreads();

    for (int row = 0; row < 16; ++row) {
        float mx = -3.4e38f;
        #pragma unroll
        for (int t = 0; t < 16; ++t) mx = fmaxf(mx, sc[row * 516 + lane + 32 * t]);
        mx = wred_max(mx);
        float se = 0.f;
        #pragma unroll
        for (int t = 0; t < 16; ++t) se += __expf(sc[row * 516 + lane + 32 * t] - mx);
        se = wred_sum(se);
        if (lane == 0) {
            const int pg = pt * 16 + row;
            const float sd = sc[row * 516 + pg];               // diagonal q == p
            dw[(b * 8 + h) * 512 + pg] = __expf(sd - mx) / se;
        }
    }
}

// ---------------------------------------------------------------------------
// K5a: attn = diag(weights) * V * sigmoid(G); the bhpd<->bNA raw reshapes
// cancel, so this is elementwise on flat [b, N*A] with a diag-weight gather.
// ---------------------------------------------------------------------------
__global__ __launch_bounds__(256) void kp_gate(
    const float* __restrict__ vf, const float* __restrict__ gf,
    const float* __restrict__ dw, unsigned short* __restrict__ attnbf)
{
    const int i = blockIdx.x * 256 + threadIdx.x;              // < 524288
    const int b = i >> 17;
    const int x = i & 131071;
    const int h = x >> 14;
    const int p = (x >> 5) & 511;
    const float w = dw[(b * 8 + h) * 512 + p];
    const float s = 1.f / (1.f + __expf(-gf[i]));
    attnbf[i] = f2bf(w * vf[i] * s);
}

// ---------------------------------------------------------------------------
// K5b: out = attn @ Wo + bo (WMMA bf16, fp32 out).
// ---------------------------------------------------------------------------
__global__ __launch_bounds__(32) void kp_out(
    const unsigned short* __restrict__ attnbf,
    const float* __restrict__ Wo, const float* __restrict__ bo,
    float* __restrict__ out)
{
    const int lane = threadIdx.x;
    const int g    = lane >> 4;
    const int m    = lane & 15;
    const int row0 = blockIdx.x * 16;
    const int col0 = blockIdx.y * 16;

    v8f acc = {0.f,0.f,0.f,0.f,0.f,0.f,0.f,0.f};
    for (int ks = 0; ks < 256; ks += 32) {
        union { v16bf v; u4 u[2]; } A;
        const unsigned short* ar = attnbf + (row0 + m) * 256 + ks;
        A.u[0] = *(const u4*)(ar + 8 * g);
        A.u[1] = *(const u4*)(ar + 16 + 8 * g);

        union { v16bf v; unsigned short s[16]; } Bf;
        #pragma unroll
        for (int e = 0; e < 16; ++e)
            Bf.s[e] = f2bf(Wo[(ks + 8 * g + (e & 7) + 16 * (e >> 3)) * 256 + col0 + m]);

        acc = __builtin_amdgcn_wmma_f32_16x16x32_bf16(
                  false, A.v, false, Bf.v, (short)0, acc, false, false);
    }
    const int col = col0 + m;
    const float bc = bo[col];
    #pragma unroll
    for (int r = 0; r < 8; ++r)
        out[(row0 + r + 8 * g) * 256 + col] = acc[r] + bc;
}

// ---------------------------------------------------------------------------
// Workspace layout (bytes): total ~42.01 MB
//   bias/Bp flat : [0,        33554432)   4*8*512*512 f32
//   Fn bf16      : [33554432, 34603008)
//   Q bf16       : [34603008, 35651584)
//   K bf16       : [35651584, 36700160)
//   V f32        : [36700160, 38797312)
//   G f32        : [38797312, 40894464)
//   diagw f32    : [40894464, 40960000)
//   attn bf16    : [40960000, 42008576)
//   W' bf16      : [42008576, 42010624)   128*8
//   S f32        : [42010624, 42010880)
//   bk f32       : [42010880, 42011136)
// ---------------------------------------------------------------------------
extern "C" void kernel_launch(void* const* d_in, const int* in_sizes, int n_in,
                              void* d_out, int out_size, void* d_ws, size_t ws_size,
                              hipStream_t stream)
{
    const float* F     = (const float*)d_in[0];
    const float* Dm    = (const float*)d_in[1];
    const float* Wq    = (const float*)d_in[2];
    const float* Wk    = (const float*)d_in[3];
    const float* Wv    = (const float*)d_in[4];
    const float* Wg    = (const float*)d_in[5];
    const float* bg    = (const float*)d_in[6];
    const float* Wp    = (const float*)d_in[7];
    const float* Wo    = (const float*)d_in[8];
    const float* bo    = (const float*)d_in[9];
    const float* gpair = (const float*)d_in[10];
    const float* bpair = (const float*)d_in[11];
    const float* gfeat = (const float*)d_in[12];
    const float* bfeat = (const float*)d_in[13];

    char* ws = (char*)d_ws;
    float*          bpb    = (float*)(ws);
    unsigned short* fnbf   = (unsigned short*)(ws + 33554432);
    unsigned short* qbf    = (unsigned short*)(ws + 34603008);
    unsigned short* kbf    = (unsigned short*)(ws + 35651584);
    float*          vf     = (float*)(ws + 36700160);
    float*          gfl    = (float*)(ws + 38797312);
    float*          dw     = (float*)(ws + 40894464);
    unsigned short* attnbf = (unsigned short*)(ws + 40960000);
    unsigned short* wpb    = (unsigned short*)(ws + 42008576);
    float*          Svec   = (float*)(ws + 42010624);
    float*          Bkvec  = (float*)(ws + 42010880);

    kp_prep  <<<1, 32, 0, stream>>>(Wp, gpair, bpair, wpb, Svec, Bkvec);
    kp_pair  <<<8192, 256, 0, stream>>>(Dm, wpb, Svec, Bkvec, bpb);
    kp_featln<<<256, 256, 0, stream>>>(F, gfeat, bfeat, fnbf);
    kp_qkvg  <<<dim3(128, 16, 4), 32, 0, stream>>>(fnbf, Wq, Wk, Wv, Wg, bg,
                                                   qbf, kbf, vf, gfl);
    kp_scores<<<dim3(32, 8, 4), 32, 0, stream>>>(qbf, kbf, bpb, dw);
    kp_gate  <<<2048, 256, 0, stream>>>(vf, gfl, dw, attnbf);
    kp_out   <<<dim3(128, 16), 32, 0, stream>>>(attnbf, Wo, bo, (float*)d_out);
}
